// GVendiVLMCriterion_61718680043972
// MI455X (gfx1250) — compile-verified, matrix operand order checked by reading, not run on hardware
//
#include <hip/hip_runtime.h>
#include <hip/hip_bf16.h>
#include <math.h>

// ---------------------------------------------------------------------------
// GVendi VLM criterion for MI455X (gfx1250, wave32, WMMA)
//
// Dominant cost: z = g @ S, (64 x 1M) x (1M x 512), S = generated +/-1 signs.
// 68.7 GFLOP vs 256 MB HBM traffic (11 us floor @ 23.3 TB/s) -> matrix pipes.
//
// Register-tiled WMMA: each wave owns a 64-row x 32-col output tile
// (4 row-tiles x 2 col-tiles = 8 f32 accumulators). K loop unrolled x2 with
// ping-pong A fragments AND both stages' sign fragments computed up front,
// so no VALU write ever lands on a WMMA source within the hazard window.
// ---------------------------------------------------------------------------

typedef __attribute__((ext_vector_type(16))) _Float16 v16h;
typedef __attribute__((ext_vector_type(8)))  float    v8f;

#define B_ROWS   64
#define D_DIM    1048576
#define PROJ_N   512
#define K_CENT   60
#define REG_EPS  0.05f
#define SK_ITERS 100
#define LAM_OT   1.2f
#define LAM_COM  1.0f
#define RAD_SEED 20u

#define RTILES   4                        // 4 x 16 = all 64 rows per wave
#define CTILES   2                        // 2 x 16 = 32 columns per wave
#define KSPLIT   128                      // K-chunks (grid.y)
#define KCHUNK   (D_DIM / KSPLIT)         // 8192 K per wave -> 128 double-steps

union AFrag { v16h v; _Float16 h[16]; };

// cheap deterministic mixer (stand-in for threefry; pure function of indices)
__device__ __forceinline__ unsigned mix32(unsigned h) {
  h ^= h >> 16; h *= 0x7FEB352Du;
  h ^= h >> 15; h *= 0x846CA68Bu;
  h ^= h >> 16;
  return h;
}

// 16 packed +/-1 f16 halves from 16 bits of h (bits khalf..khalf+15).
// +1.0h = 0x3C00, -1.0h = 0xBC00: XOR the sign bit in, branch-free.
__device__ __forceinline__ v16h sign_frag(unsigned h, int khalf) {
  const unsigned hh = h >> khalf;
  union { v16h v; unsigned u[8]; } b;
#pragma unroll
  for (int i = 0; i < 8; ++i) {
    b.u[i] = 0x3C003C00u ^ ((hh << (15 - 2 * i)) & 0x8000u)
                         ^ ((hh << (30 - 2 * i)) & 0x80000000u);
  }
  return b.v;
}

// load 16 contiguous floats (four b128 loads) and convert to f16
__device__ __forceinline__ void load_cvt16(const float* __restrict__ p, _Float16* h) {
  const float4* q = reinterpret_cast<const float4*>(p);
#pragma unroll
  for (int i = 0; i < 4; ++i) {
    float4 f = q[i];
    h[4 * i + 0] = (_Float16)f.x;
    h[4 * i + 1] = (_Float16)f.y;
    h[4 * i + 2] = (_Float16)f.z;
    h[4 * i + 3] = (_Float16)f.w;
  }
}

// A fragments for all 4 row tiles at K position kb
__device__ __forceinline__ void load_afrags(const float* __restrict__ g0,
                                            int kb, int khalf, AFrag* a) {
#pragma unroll
  for (int r = 0; r < RTILES; ++r)
    load_cvt16(g0 + (size_t)(r * 16) * D_DIM + kb + khalf, a[r].h);
}

// sign fragments for one 32-K step (one per column tile)
__device__ __forceinline__ void make_bfrags(int kb, const unsigned* colterm,
                                            int khalf, v16h* bfrag) {
  const unsigned kterm = ((unsigned)(kb >> 5) ^ RAD_SEED) * 0x9E3779B1u;
#pragma unroll
  for (int c = 0; c < CTILES; ++c)
    bfrag[c] = sign_frag(mix32(kterm ^ colterm[c]), khalf);
}

// 8 independent WMMAs for one 32-K step
__device__ __forceinline__ void wmma_step(const AFrag* a, const v16h* bfrag,
                                          v8f acc[RTILES][CTILES]) {
#pragma unroll
  for (int r = 0; r < RTILES; ++r)
#pragma unroll
    for (int c = 0; c < CTILES; ++c)
      acc[r][c] = __builtin_amdgcn_wmma_f32_16x16x32_f16(
          false, a[r].v, false, bfrag[c], (short)0, acc[r][c], false, false);
}

__global__ void zero_kernel(float* __restrict__ p, int n) {
  int i = blockIdx.x * blockDim.x + threadIdx.x;
  if (i < n) p[i] = 0.0f;
}

// ---------------------------------------------------------------------------
// Kernel 1: projection partials. grid = (PROJ/32, KSPLIT), block = 32.
// A frag: lane&15 = M, lane bit4 selects K-half. B frag: lane&15 = N, same.
// C/D frag: VGPR r -> M = r + 8*(lane>=16), N = lane&15.
// ---------------------------------------------------------------------------
__global__ __launch_bounds__(32)
void proj_wmma_kernel(const float* __restrict__ g, float* __restrict__ z) {
  const int lane    = threadIdx.x;
  const int m       = lane & 15;
  const int khalf   = lane & 16;               // 0 or 16
  const int colBase = blockIdx.x * (16 * CTILES);
  const int k0      = blockIdx.y * KCHUNK;
  const int kEnd    = k0 + KCHUNK;

  // per-column hash terms (hoisted out of the K loop)
  unsigned colterm[CTILES];
#pragma unroll
  for (int c = 0; c < CTILES; ++c)
    colterm[c] = (unsigned)(colBase + 16 * c + m) * 0x85EBCA77u ^ 0x27220A95u;

  const float* __restrict__ g0 = g + (size_t)m * D_DIM;   // row m of each tile

  v8f acc[RTILES][CTILES];
#pragma unroll
  for (int r = 0; r < RTILES; ++r)
#pragma unroll
    for (int c = 0; c < CTILES; ++c)
      acc[r][c] = {};

  // explicit double-buffered software pipeline over 64-K double-steps
  AFrag aA[RTILES], aB[RTILES];
  load_afrags(g0, k0, khalf, aA);

  for (int kb = k0; kb < kEnd; kb += 64) {
    // both stages' sign fragments live BEFORE any WMMA issues: no VALU write
    // can land on a WMMA B-source inside the hazard window.
    v16h bf0[CTILES], bf1[CTILES];
    make_bfrags(kb, colterm, khalf, bf0);
    make_bfrags(kb + 32, colterm, khalf, bf1);

    {
      const int pr = (kb >> 5) & (RTILES - 1);   // rotate prefetch over rows
      __builtin_prefetch(g0 + (size_t)(pr * 16) * D_DIM + kb + 256, 0, 1);
    }

    // stage 0: consume aA, fill aB
    load_afrags(g0, kb + 32, khalf, aB);
    wmma_step(aA, bf0, acc);

    // stage 1: consume aB, fill aA (clamped dummy load on the final step)
    const int kn = (kb + 64 < kEnd) ? (kb + 64) : k0;
    load_afrags(g0, kn, khalf, aA);
    wmma_step(aB, bf1, acc);
  }

  // merge partials: z[row][col] += acc * 512^-0.5
  const float scale = 0.044194173824159216f;
  const int mOff = (lane & 16) ? 8 : 0;
#pragma unroll
  for (int r = 0; r < RTILES; ++r)
#pragma unroll
    for (int c = 0; c < CTILES; ++c) {
      const int col = colBase + 16 * c + (lane & 15);
#pragma unroll
      for (int rr = 0; rr < 8; ++rr)
        atomicAdd(&z[(size_t)(r * 16 + mOff + rr) * PROJ_N + col],
                  acc[r][c][rr] * scale);
    }
}

// ---------------------------------------------------------------------------
// Kernel 2: e = z @ W_pj^T  (64x512 @ 512x512), f16 WMMA, 16 steps per wave.
// grid = (PROJ/16, B/16), block = 32. e[b][p] = dot(z[b,:], W[p,:]).
// ---------------------------------------------------------------------------
__global__ __launch_bounds__(32)
void wproj_wmma_kernel(const float* __restrict__ z, const float* __restrict__ W,
                       float* __restrict__ e) {
  const int lane    = threadIdx.x;
  const int colTile = blockIdx.x;
  const int rowTile = blockIdx.y;

  const int m     = lane & 15;
  const int khalf = lane & 16;
  const int arow  = rowTile * 16 + m;
  const int bcol  = colTile * 16 + m;      // output column p -> W row p

  v8f acc = {};
  for (int kb = 0; kb < PROJ_N; kb += 32) {
    AFrag a, b;
    load_cvt16(z + (size_t)arow * PROJ_N + kb + khalf, a.h);
    load_cvt16(W + (size_t)bcol * PROJ_N + kb + khalf, b.h);
    acc = __builtin_amdgcn_wmma_f32_16x16x32_f16(
        false, a.v, false, b.v, (short)0, acc, false, false);
  }

  const int mBase = rowTile * 16 + ((lane & 16) ? 8 : 0);
  const int col   = colTile * 16 + (lane & 15);
#pragma unroll
  for (int r = 0; r < 8; ++r)
    e[(size_t)(mBase + r) * PROJ_N + col] = acc[r];
}

// ---------------------------------------------------------------------------
// Kernel 3: normalize centroids, C = sqdist(e, cb), 100 log-Sinkhorn iters,
// loss = 1.2*sum(T*C) + mean(min_k C).  Single 256-thread workgroup, LDS-held.
// ---------------------------------------------------------------------------
__global__ __launch_bounds__(256)
void sinkhorn_kernel(const float* __restrict__ e, const float* __restrict__ cent,
                     float* __restrict__ out) {
  __shared__ float C[B_ROWS][K_CENT];
  __shared__ float lu[B_ROWS];
  __shared__ float lv[K_CENT];
  __shared__ float invn[K_CENT];
  __shared__ float red[256];

  const int t = threadIdx.x;
  const float invReg = 1.0f / REG_EPS;

  if (t < K_CENT) {
    float s = 0.0f;
    for (int d = 0; d < PROJ_N; ++d) {
      float x = cent[(size_t)t * PROJ_N + d];
      s += x * x;
    }
    invn[t] = 1.0f / fmaxf(sqrtf(s), 1e-12f);
  }
  __syncthreads();

  for (int idx = t; idx < B_ROWS * K_CENT; idx += 256) {
    const int bi = idx / K_CENT, k = idx % K_CENT;
    const float iv = invn[k];
    float ne = 0.0f, nc = 0.0f, dot = 0.0f;
    for (int d = 0; d < PROJ_N; ++d) {
      float ev = e[(size_t)bi * PROJ_N + d];
      float cv = cent[(size_t)k * PROJ_N + d] * iv;
      ne += ev * ev; nc += cv * cv; dot += ev * cv;
    }
    C[bi][k] = fmaxf(ne + nc - 2.0f * dot, 0.0f);
  }
  if (t < B_ROWS) lu[t] = 0.0f;
  if (t < K_CENT) lv[t] = 0.0f;
  __syncthreads();

  const float log_a = -logf((float)B_ROWS);
  const float log_b = -logf((float)K_CENT);

  for (int it = 0; it < SK_ITERS; ++it) {
    if (t < B_ROWS) {                          // lu = log_a - lse_k(lv - M)
      float mx = -INFINITY;
      for (int k = 0; k < K_CENT; ++k)
        mx = fmaxf(mx, lv[k] - C[t][k] * invReg);
      float s = 0.0f;
      for (int k = 0; k < K_CENT; ++k)
        s += expf(lv[k] - C[t][k] * invReg - mx);
      lu[t] = log_a - (mx + logf(s));
    }
    __syncthreads();
    if (t < K_CENT) {                          // lv = log_b - lse_b(lu - M)
      float mx = -INFINITY;
      for (int bi = 0; bi < B_ROWS; ++bi)
        mx = fmaxf(mx, lu[bi] - C[bi][t] * invReg);
      float s = 0.0f;
      for (int bi = 0; bi < B_ROWS; ++bi)
        s += expf(lu[bi] - C[bi][t] * invReg - mx);
      lv[t] = log_b - (mx + logf(s));
    }
    __syncthreads();
  }

  float part = 0.0f;
  for (int idx = t; idx < B_ROWS * K_CENT; idx += 256) {
    const int bi = idx / K_CENT, k = idx % K_CENT;
    const float Cv = C[bi][k];
    part += expf(lu[bi] + lv[k] - Cv * invReg) * Cv;
  }
  float commit = 0.0f;
  if (t < B_ROWS) {
    float mn = INFINITY;
    for (int k = 0; k < K_CENT; ++k) mn = fminf(mn, C[t][k]);
    commit = mn;
  }
  red[t] = LAM_OT * part + LAM_COM * commit * (1.0f / (float)B_ROWS);
  __syncthreads();
  for (int s = 128; s > 0; s >>= 1) {
    if (t < s) red[t] += red[t + s];
    __syncthreads();
  }
  if (t == 0) out[0] = red[0];
}

// ---------------------------------------------------------------------------
extern "C" void kernel_launch(void* const* d_in, const int* in_sizes, int n_in,
                              void* d_out, int out_size, void* d_ws, size_t ws_size,
                              hipStream_t stream) {
  const float* g    = (const float*)d_in[0];   // [64, 1048576]
  const float* W_pj = (const float*)d_in[1];   // [512, 512]
  const float* cent = (const float*)d_in[2];   // [60, 512]
  float* out = (float*)d_out;

  float* z = (float*)d_ws;                             // [64, 512]
  float* e = z + (size_t)B_ROWS * PROJ_N;              // [64, 512]

  const int zn = B_ROWS * PROJ_N;                      // 32768
  zero_kernel<<<(zn + 255) / 256, 256, 0, stream>>>(z, zn);

  dim3 pgrid(PROJ_N / (16 * CTILES), KSPLIT);          // (16, 128)
  proj_wmma_kernel<<<pgrid, 32, 0, stream>>>(g, z);

  dim3 wgrid(PROJ_N / 16, B_ROWS / 16);                // (32, 4)
  wproj_wmma_kernel<<<wgrid, 32, 0, stream>>>(z, W_pj, e);

  sinkhorn_kernel<<<1, 256, 0, stream>>>(e, cent, out);
}
